// DynamicsStream_44152263803571
// MI455X (gfx1250) — matrix operand order repaired and checked by reading, not compiled, not run on hardware
//
#include <hip/hip_runtime.h>

#define NWG   8
#define NTHR  256
#define T_LEN 2048
#define B_SZ  32
#define IN_F  256
#define H_F   512
#define OUT_F 256
#define PAD   8
#define S_IN  (IN_F + PAD)   // 264 halfs per row
#define S_H   (H_F + PAD)    // 520 halfs per row
#define DT_C  0.1f

typedef __attribute__((ext_vector_type(16))) __bf16 v16bf;
typedef __attribute__((ext_vector_type(8)))  float  v8f;
typedef __attribute__((ext_vector_type(4)))  float  f4;
typedef __attribute__((ext_vector_type(4)))  unsigned int u4;
typedef __attribute__((ext_vector_type(2)))  unsigned int u2;

union BFrag { v16bf v; u4 q[2]; unsigned short h[16]; };

__device__ __forceinline__ unsigned short f2bf(float f) {
  unsigned u = __float_as_uint(f);
  u += 0x7FFFu + ((u >> 16) & 1u);           // round-to-nearest-even
  return (unsigned short)(u >> 16);
}

// B fragment (K=32 x N=16 bf16): lane holds K-values [8h,8h+8) and [16+8h,16+8h+8) of one column.
__device__ __forceinline__ v16bf lds_bfrag(const unsigned short* w, int stride,
                                           int nrow, int k0, int half) {
  const unsigned short* p = w + nrow * stride + k0 + half * 8;
  BFrag f;
  f.q[0] = *(const u4*)(p);
  f.q[1] = *(const u4*)(p + 16);
  return f.v;
}

// A fragment (M=16 x K=32 bf16) from row-major bf16, given flat row offset (halfs).
__device__ __forceinline__ v16bf gA_bf_off(const unsigned short* g, size_t rowoff,
                                           int k0, int half) {
  const unsigned short* p = g + rowoff + k0 + half * 8;
  BFrag f;
  f.q[0] = *(const u4*)(p);
  f.q[1] = *(const u4*)(p + 16);
  return f.v;
}

// Fallback: A fragment from fp32 global with in-register bf16 conversion.
__device__ __forceinline__ v16bf gA_f32(const float* x, size_t rowoff, int k0, int half) {
  const float* p = x + rowoff + k0 + half * 8;
  f4 a0 = *(const f4*)(p);
  f4 a1 = *(const f4*)(p + 4);
  f4 b0 = *(const f4*)(p + 16);
  f4 b1 = *(const f4*)(p + 20);
  BFrag f;
#pragma unroll
  for (int i = 0; i < 4; ++i) {
    f.h[i]      = f2bf(a0[i]);
    f.h[4 + i]  = f2bf(a1[i]);
    f.h[8 + i]  = f2bf(b0[i]);
    f.h[12 + i] = f2bf(b1[i]);
  }
  return f.v;
}

#define WMMA_BF16(A, B, C) \
  __builtin_amdgcn_wmma_f32_16x16x32_bf16(false, (A), false, (B), (short)0, (C), false, false)

// Monotonic-counter device-wide barrier (8 blocks, all co-resident).
__device__ __forceinline__ void grid_barrier(unsigned* bar, unsigned target) {
  __syncthreads();
  __threadfence();                        // release
  if (threadIdx.x == 0) {
    atomicAdd(bar, 1u);
    while (__hip_atomic_load(bar, __ATOMIC_RELAXED, __HIP_MEMORY_SCOPE_AGENT) < target) {
      __builtin_amdgcn_s_sleep(1);
    }
  }
  __syncthreads();
  __threadfence();                        // acquire
}

// One-time prepass: convert x (fp32) -> bf16 copy in scratch. Pure-bandwidth.
extern "C" __global__ void xconv_kernel(const float* __restrict__ x,
                                        unsigned short* __restrict__ xb, int n4) {
  int i = blockIdx.x * blockDim.x + threadIdx.x;
  const int stride = gridDim.x * blockDim.x;
  for (; i < n4; i += stride) {
    f4 v = ((const f4*)x)[i];
    u2 o;
    o.x = (unsigned)f2bf(v.x) | ((unsigned)f2bf(v.y) << 16);
    o.y = (unsigned)f2bf(v.z) | ((unsigned)f2bf(v.w) << 16);
    ((u2*)xb)[i] = o;
  }
}

extern "C" __global__ __launch_bounds__(NTHR, 1)
void ctrnn_persistent(const float* __restrict__ x,
                      const float* __restrict__ Win0, const float* __restrict__ bin0,
                      const float* __restrict__ Wrec0, const float* __restrict__ tau0,
                      const float* __restrict__ Win1, const float* __restrict__ bin1,
                      const float* __restrict__ Wrec1, const float* __restrict__ tau1,
                      const float* __restrict__ Wout, const float* __restrict__ bo_p,
                      float* __restrict__ out, unsigned char* __restrict__ ws,
                      const unsigned short* __restrict__ xb) {
  extern __shared__ unsigned short smem[];
  const int wg  = blockIdx.x;             // owns h columns [wg*64, wg*64+64), out cols [wg*32, +32)
  const int tid = threadIdx.x;

  // LDS layout (halfs), rows padded by 8 halfs to kill ds bank conflicts
  const int oWin0  = 0;
  const int oWrec0 = oWin0  + 64 * S_IN;
  const int oWin1  = oWrec0 + 64 * S_H;
  const int oWrec1 = oWin1  + 64 * S_H;
  const int oWout  = oWrec1 + 64 * S_H;   // total 133376 halfs = 260.5 KB

  // ---- one-time weight preload: fp32 global -> bf16 LDS (this WG's column slice) ----
  for (int i = tid; i < 64 * IN_F; i += NTHR) {
    int n = i / IN_F, k = i - n * IN_F;
    smem[oWin0 + n * S_IN + k] = f2bf(Win0[(size_t)(wg * 64 + n) * IN_F + k]);
  }
  for (int i = tid; i < 64 * H_F; i += NTHR) {
    int n = i / H_F, k = i - n * H_F;
    size_t g = (size_t)(wg * 64 + n) * H_F + k;
    smem[oWrec0 + n * S_H + k] = f2bf(Wrec0[g]);
    smem[oWin1  + n * S_H + k] = f2bf(Win1[g]);
    smem[oWrec1 + n * S_H + k] = f2bf(Wrec1[g]);
  }
  for (int i = tid; i < 32 * H_F; i += NTHR) {
    int n = i / H_F, k = i - n * H_F;
    smem[oWout + n * S_H + k] = f2bf(Wout[(size_t)(wg * 32 + n) * H_F + k]);
  }
  __syncthreads();

  unsigned*       bar = (unsigned*)ws;
  unsigned short* h0b = (unsigned short*)(ws + 64);                          // 2 x [32][512] bf16
  unsigned short* h1b = (unsigned short*)(ws + 64 + 2 * B_SZ * H_F * 2);     // 2 x [32][512] bf16

  const int lane = tid & 31;
  const int wave = tid >> 5;
  const int half = lane >> 4;
  const int lcol = lane & 15;

  // hidden-layer tile mapping: 2 M-tiles x 4 N-tiles = 8 waves
  const int mt    = wave & 1;
  const int ntl   = wave >> 1;                  // 0..3
  const int mrow  = mt * 16 + lcol;             // A-fragment row (batch index)
  const int nrow  = ntl * 16 + lcol;            // B row within 64-col slice
  const int ncol  = wg * 64 + nrow;             // global hidden column
  const int mbase = mt * 16 + half * 8;         // first C/D row for this lane

  // output tile mapping: 2 M-tiles x 2 N-tiles = waves 0..3
  const int mtc    = wave & 1;
  const int ntc    = (wave >> 1) & 1;
  const int mrowc  = mtc * 16 + lcol;
  const int nrowc  = ntc * 16 + lcol;
  const int ncolc  = wg * 32 + nrowc;
  const int mbasec = mtc * 16 + half * 8;

  const float bi0 = bin0[ncol];
  const float r0  = DT_C / fminf(fmaxf(tau0[ncol], 0.1f), 10.0f);   // dt / clamp(tau)
  const float bi1 = bin1[ncol];
  const float r1  = DT_C / fminf(fmaxf(tau1[ncol], 0.1f), 10.0f);
  const float bov = bo_p[ncolc];

  const bool use_xb = (xb != nullptr);

  const v8f vzero = {0.f, 0.f, 0.f, 0.f, 0.f, 0.f, 0.f, 0.f};
  v8f h0s = vzero;                              // fp32 recurrent state lives in VGPRs
  v8f h1s = vzero;

  unsigned bcount = 0;

  for (int t = 0; t < T_LEN; ++t) {
    const int p = t & 1;

    // ================= layer 0: x@Win0^T + h0@Wrec0^T (2 indep WMMA chains) ==========
    v8f accA = vzero, accB = vzero;
    const size_t xrow = ((size_t)mrow * T_LEN + t) * IN_F;
    if (use_xb) {
      __builtin_prefetch(xb + xrow + IN_F, 0, 1);       // next timestep's x row
#pragma unroll
      for (int kt = 0; kt < IN_F / 32; ++kt) {
        v16bf a = gA_bf_off(xb, xrow, kt * 32, half);
        v16bf b = lds_bfrag(smem + oWin0, S_IN, nrow, kt * 32, half);
        accA = WMMA_BF16(a, b, accA);
      }
    } else {
      __builtin_prefetch(x + xrow + IN_F, 0, 1);
#pragma unroll
      for (int kt = 0; kt < IN_F / 32; ++kt) {
        v16bf a = gA_f32(x, xrow, kt * 32, half);
        v16bf b = lds_bfrag(smem + oWin0, S_IN, nrow, kt * 32, half);
        accA = WMMA_BF16(a, b, accA);
      }
    }
    const unsigned short* h0r = h0b + (size_t)p * (B_SZ * H_F);
#pragma unroll 4
    for (int kt = 0; kt < H_F / 32; ++kt) {
      v16bf a = gA_bf_off(h0r, (size_t)mrow * H_F, kt * 32, half);
      v16bf b = lds_bfrag(smem + oWrec0, S_H, nrow, kt * 32, half);
      accB = WMMA_BF16(a, b, accB);
    }
    v8f acc = accA + accB;
    unsigned short* h0w = h0b + (size_t)(p ^ 1) * (B_SZ * H_F);
#pragma unroll
    for (int j = 0; j < 8; ++j) {
      float targ = tanhf(acc[j] + bi0);
      float hn = h0s[j] + (targ - h0s[j]) * r0;
      h0s[j] = hn;
      h0w[(mbase + j) * H_F + ncol] = f2bf(hn);
    }
    grid_barrier(bar, ++bcount * NWG);          // h0_new visible everywhere

    // ================= layer 1: h0n@Win1^T + h1@Wrec1^T (2 indep WMMA chains) ========
    accA = vzero; accB = vzero;
    const unsigned short* h0n = h0b + (size_t)(p ^ 1) * (B_SZ * H_F);
    const unsigned short* h1r = h1b + (size_t)p * (B_SZ * H_F);
#pragma unroll 4
    for (int kt = 0; kt < H_F / 32; ++kt) {
      v16bf a = gA_bf_off(h0n, (size_t)mrow * H_F, kt * 32, half);
      v16bf b = lds_bfrag(smem + oWin1, S_H, nrow, kt * 32, half);
      accA = WMMA_BF16(a, b, accA);
    }
#pragma unroll 4
    for (int kt = 0; kt < H_F / 32; ++kt) {
      v16bf a = gA_bf_off(h1r, (size_t)mrow * H_F, kt * 32, half);
      v16bf b = lds_bfrag(smem + oWrec1, S_H, nrow, kt * 32, half);
      accB = WMMA_BF16(a, b, accB);
    }
    acc = accA + accB;
    unsigned short* h1w = h1b + (size_t)(p ^ 1) * (B_SZ * H_F);
#pragma unroll
    for (int j = 0; j < 8; ++j) {
      float targ = tanhf(acc[j] + bi1);
      float hn = h1s[j] + (targ - h1s[j]) * r1;
      h1s[j] = hn;
      h1w[(mbase + j) * H_F + ncol] = f2bf(hn);
    }
    grid_barrier(bar, ++bcount * NWG);          // h1_new visible everywhere

    // ================= output projection + residual (even/odd K chains) =============
    if (wave < 4) {                             // wave-uniform branch: EXEC stays all-ones
      accA = vzero; accB = vzero;
      const unsigned short* h1n = h1b + (size_t)(p ^ 1) * (B_SZ * H_F);
#pragma unroll 4
      for (int kt = 0; kt < H_F / 64; ++kt) {
        v16bf a0 = gA_bf_off(h1n, (size_t)mrowc * H_F, kt * 64, half);
        v16bf b0 = lds_bfrag(smem + oWout, S_H, nrowc, kt * 64, half);
        accA = WMMA_BF16(a0, b0, accA);
        v16bf a1 = gA_bf_off(h1n, (size_t)mrowc * H_F, kt * 64 + 32, half);
        v16bf b1 = lds_bfrag(smem + oWout, S_H, nrowc, kt * 64 + 32, half);
        accB = WMMA_BF16(a1, b1, accB);
      }
      acc = accA + accB;
#pragma unroll
      for (int j = 0; j < 8; ++j) {
        size_t o = ((size_t)(mbasec + j) * T_LEN + t) * OUT_F + ncolc;
        out[o] = acc[j] + bov + x[o];           // residual: IN_F == OUT_F, same layout
      }
    }
    // no barrier needed: next step's writes touch opposite-parity buffers,
    // and barrier1 of step t+1 orders them after this phase.
  }
}

extern "C" void kernel_launch(void* const* d_in, const int* in_sizes, int n_in,
                              void* d_out, int out_size, void* d_ws, size_t ws_size,
                              hipStream_t stream) {
  (void)in_sizes; (void)n_in; (void)out_size;
  const float* x     = (const float*)d_in[0];
  const float* Win0  = (const float*)d_in[1];
  const float* bin0  = (const float*)d_in[2];
  const float* Wrec0 = (const float*)d_in[3];
  const float* tau0  = (const float*)d_in[4];
  const float* Win1  = (const float*)d_in[5];
  const float* bin1  = (const float*)d_in[6];
  const float* Wrec1 = (const float*)d_in[7];
  const float* tau1  = (const float*)d_in[8];
  const float* Wout  = (const float*)d_in[9];
  const float* bo    = (const float*)d_in[10];

  // zero barrier counter + both h double-buffers (bf16 zero == 0x0000)
  const size_t clear_bytes = 64 + 4 * (size_t)B_SZ * H_F * 2;
  hipMemsetAsync(d_ws, 0, clear_bytes, stream);

  // optional prepass: one-time fp32 -> bf16 copy of x into scratch
  const size_t XOFF   = 262144;                                   // 256 KB
  const size_t xbytes = (size_t)B_SZ * T_LEN * IN_F * 2;          // 33.5 MB
  unsigned short* xbf = nullptr;
  if (ws_size >= XOFF + xbytes) {
    xbf = (unsigned short*)((char*)d_ws + XOFF);
    const int n4 = (B_SZ * T_LEN * IN_F) / 4;
    hipLaunchKernelGGL(xconv_kernel, dim3(2048), dim3(256), 0, stream, x, xbf, n4);
  }

  const size_t lds_bytes = (size_t)(64 * S_IN + 3 * 64 * S_H + 32 * S_H) * 2;  // 266752 B

  (void)hipFuncSetAttribute((const void*)ctrnn_persistent,
                            hipFuncAttributeMaxDynamicSharedMemorySize,
                            (int)lds_bytes);

  hipLaunchKernelGGL(ctrnn_persistent, dim3(NWG), dim3(NTHR), lds_bytes, stream,
                     x, Win0, bin0, Wrec0, tau0, Win1, bin1, Wrec1, tau1, Wout, bo,
                     (float*)d_out, (unsigned char*)d_ws, (const unsigned short*)xbf);
}
